// MoE_15221364097266
// MI455X (gfx1250) — matrix operand order repaired and checked by reading, not compiled, not run on hardware
//
#include <hip/hip_runtime.h>
#include <hip/hip_bf16.h>
#include <stdint.h>

// Problem constants (B,T,D,E,H) = (4,2048,1024,8,1024)
#define BB 4
#define TT 2048
#define DD 1024
#define EE 8
#define HH 1024
#define BT (BB * TT)          // 8192 tokens
#define M_TILE 64             // tokens per block in expert kernel (4 WMMA m-tiles)
#define NMT 4                 // m-tiles per wave
#define XS_STRIDE 1032        // padded LDS row stride (bf16 elems) to spread banks

typedef __attribute__((ext_vector_type(16))) __bf16 v16bf;
typedef __attribute__((ext_vector_type(8)))  float  v8f;

union FragBF { uint4 q[2]; v16bf v; };

__device__ __forceinline__ unsigned short f2bf(float f) {
  // round-to-nearest-even fp32 -> bf16 (raw bits)
  unsigned int u = __float_as_uint(f);
  u += 0x7FFFu + ((u >> 16) & 1u);
  return (unsigned short)(u >> 16);
}

__device__ __forceinline__ void load_b(const unsigned short* p, FragBF& b) {
  const uint4* q = (const uint4*)p;
  b.q[0] = q[0];
  b.q[1] = q[2];
}

#define WMMA_BF16(A, B, C) \
  __builtin_amdgcn_wmma_f32_16x16x32_bf16(false, (A), false, (B), (short)0, (C), false, false)

// One 32-K step of the dual (gate,up) GEMM: 4 A-tiles x 2 B matrices = 8 WMMAs.
// All A loads are immediate-offset ds_loads off two fixed base registers.
__device__ __forceinline__ void wmma_step2(const unsigned short* __restrict__ aLo,
                                           const unsigned short* __restrict__ aHi,
                                           int k,
                                           const FragBF& bg, const FragBF& bu,
                                           v8f* accg, v8f* accu) {
  FragBF f0, f1, f2, f3;
  f0.q[0] = *(const uint4*)(aLo + k);
  f0.q[1] = *(const uint4*)(aLo + k + 16);
  f1.q[0] = *(const uint4*)(aLo + 16 * XS_STRIDE + k);
  f1.q[1] = *(const uint4*)(aLo + 16 * XS_STRIDE + k + 16);
  f2.q[0] = *(const uint4*)(aHi + k);
  f2.q[1] = *(const uint4*)(aHi + k + 16);
  f3.q[0] = *(const uint4*)(aHi + 16 * XS_STRIDE + k);
  f3.q[1] = *(const uint4*)(aHi + 16 * XS_STRIDE + k + 16);
  accg[0] = WMMA_BF16(f0.v, bg.v, accg[0]);
  accu[0] = WMMA_BF16(f0.v, bu.v, accu[0]);
  accg[1] = WMMA_BF16(f1.v, bg.v, accg[1]);
  accu[1] = WMMA_BF16(f1.v, bu.v, accu[1]);
  accg[2] = WMMA_BF16(f2.v, bg.v, accg[2]);
  accu[2] = WMMA_BF16(f2.v, bu.v, accu[2]);
  accg[3] = WMMA_BF16(f3.v, bg.v, accg[3]);
  accu[3] = WMMA_BF16(f3.v, bu.v, accu[3]);
}

// Single-B 32-K step (down projection): 4 WMMAs
__device__ __forceinline__ void wmma_step1(const unsigned short* __restrict__ aLo,
                                           const unsigned short* __restrict__ aHi,
                                           int k,
                                           const FragBF& bd, v8f* acc) {
  FragBF f0, f1, f2, f3;
  f0.q[0] = *(const uint4*)(aLo + k);
  f0.q[1] = *(const uint4*)(aLo + k + 16);
  f1.q[0] = *(const uint4*)(aLo + 16 * XS_STRIDE + k);
  f1.q[1] = *(const uint4*)(aLo + 16 * XS_STRIDE + k + 16);
  f2.q[0] = *(const uint4*)(aHi + k);
  f2.q[1] = *(const uint4*)(aHi + k + 16);
  f3.q[0] = *(const uint4*)(aHi + 16 * XS_STRIDE + k);
  f3.q[1] = *(const uint4*)(aHi + 16 * XS_STRIDE + k + 16);
  acc[0] = WMMA_BF16(f0.v, bd.v, acc[0]);
  acc[1] = WMMA_BF16(f1.v, bd.v, acc[1]);
  acc[2] = WMMA_BF16(f2.v, bd.v, acc[2]);
  acc[3] = WMMA_BF16(f3.v, bd.v, acc[3]);
}

// ---------------------------------------------------------------------------
// 1) Routing: logits = x . Wg[e], argmax, build per-expert token lists
// ---------------------------------------------------------------------------
__global__ void __launch_bounds__(256) route_kernel(const float* __restrict__ x,
                                                    const float* __restrict__ Wg,
                                                    int* __restrict__ counts,
                                                    int* __restrict__ tokList) {
  const int t = blockIdx.x * 256 + threadIdx.x;   // token id, grid sized to BT
  const float4* xr = (const float4*)(x + (size_t)t * DD);
  float acc[EE];
#pragma unroll
  for (int e = 0; e < EE; ++e) acc[e] = 0.0f;
  for (int i = 0; i < DD / 4; ++i) {
    float4 xv = xr[i];
#pragma unroll
    for (int e = 0; e < EE; ++e) {
      float4 wv = ((const float4*)(Wg + (size_t)e * DD))[i];
      acc[e] += xv.x * wv.x + xv.y * wv.y + xv.z * wv.z + xv.w * wv.w;
    }
  }
  int best = 0;
  float bv = acc[0];
#pragma unroll
  for (int e = 1; e < EE; ++e)
    if (acc[e] > bv) { bv = acc[e]; best = e; }   // strict '>' == first-max (argmax tie rule)
  int pos = atomicAdd(&counts[best], 1);
  tokList[(size_t)best * BT + pos] = t;
}

// ---------------------------------------------------------------------------
// 2) Weight transpose + fp32->bf16 convert: in [E][R][C] f32 -> out [E][C][R] bf16
// ---------------------------------------------------------------------------
__global__ void transpose_cvt_kernel(const float* __restrict__ in,
                                     unsigned short* __restrict__ out,
                                     int R, int C) {
  __shared__ unsigned short tile[32][33];
  const int e  = blockIdx.z;
  const int c0 = blockIdx.x * 32, r0 = blockIdx.y * 32;
  const float* src = in + (size_t)e * R * C;
  unsigned short* dst = out + (size_t)e * R * C;
  const int tx = threadIdx.x, ty = threadIdx.y;
  tile[ty][tx] = f2bf(src[(size_t)(r0 + ty) * C + (c0 + tx)]);
  __syncthreads();
  dst[(size_t)(c0 + ty) * R + (r0 + tx)] = tile[tx][ty];
}

// ---------------------------------------------------------------------------
// 3) Fused expert kernel: per (expert, 64-token tile)
//    Phase 1: h = silu(x@G) * (x@U)   (K=D=1024, N=H=1024)
//    Phase 2: y = h @ Dn              (K=H=1024, N=D=1024)
//    8 waves, each owns a 128-wide N slice; B frags double-buffered with a
//    full-iteration lookahead; A loads are imm-offset ds_loads off lo/hi bases
//    (hi base pinned in a VGPR via an opaque asm to stop rematerialization).
// ---------------------------------------------------------------------------
__global__ void __launch_bounds__(256) moe_expert_kernel(
    const float* __restrict__ x,
    const unsigned short* __restrict__ gate_t,   // [E][H][D] bf16
    const unsigned short* __restrict__ up_t,     // [E][H][D] bf16
    const unsigned short* __restrict__ down_t,   // [E][D][H] bf16
    const int* __restrict__ counts,
    const int* __restrict__ tokList,
    float* __restrict__ y) {
  extern __shared__ unsigned short smem[];
  unsigned short* xs = smem;                          // 64 x XS_STRIDE bf16 (x tile)
  unsigned short* hs = smem + M_TILE * XS_STRIDE;     // 64 x XS_STRIDE bf16 (h tile)
  __shared__ int stok[M_TILE];
  __shared__ int scnt;

  const int e = blockIdx.x;
  const int tileBase = blockIdx.y * M_TILE;
  if (threadIdx.x == 0) scnt = counts[e];
  __syncthreads();
  const int cnt = scnt;
  if (tileBase >= cnt) return;                        // uniform exit

  if (threadIdx.x < M_TILE) {
    int gi = tileBase + threadIdx.x;
    stok[threadIdx.x] = (gi < cnt) ? tokList[(size_t)e * BT + gi] : 0;
  }
  __syncthreads();

  // Stage gathered x rows into LDS as bf16 (coalesced float4 reads)
  for (int i = threadIdx.x; i < M_TILE * (DD / 4); i += 256) {
    int row = i >> 8;                   // DD/4 == 256 float4 per row
    int c4  = i & 255;
    float4 v = ((const float4*)(x + (size_t)stok[row] * DD))[c4];
    unsigned short* p = xs + row * XS_STRIDE + c4 * 4;
    p[0] = f2bf(v.x); p[1] = f2bf(v.y); p[2] = f2bf(v.z); p[3] = f2bf(v.w);
  }
  __syncthreads();

  const int lane = threadIdx.x & 31;
  const int wv   = threadIdx.x >> 5;    // wave 0..7 -> N slice [wv*128, wv*128+128)
  const int n16  = lane & 15;           // N within 16-col tile / A row within tile
  const int kh   = lane >> 4;           // K-half select (ISA 16-bit A/B layouts)

  // Per-lane A bases: rows 0-31 (lo) and rows 32-63 (hi).  The hi offsets are
  // made opaque so the compiler keeps them live in VGPRs instead of
  // rematerializing base+66KB constants inside the k-loops (which caused
  // WMMA->VALU WAR hazard nops).
  unsigned loOff = (unsigned)(n16 * XS_STRIDE + kh * 8);
  unsigned hiOffA = loOff + 32u * XS_STRIDE;
  unsigned hiOffH = loOff + 32u * XS_STRIDE;
  asm("" : "+v"(hiOffA));
  asm("" : "+v"(hiOffH));
  const unsigned short* aLo = xs + loOff;
  const unsigned short* aHi = xs + hiOffA;
  const unsigned short* hLo = hs + loOff;
  const unsigned short* hHi = hs + hiOffH;

  // -------- Phase 1: gate & up GEMMs + SiLU --------
#pragma unroll 1
  for (int nt = 0; nt < 8; ++nt) {
    const int ncol = wv * 128 + nt * 16;
    v8f accg[NMT], accu[NMT];
#pragma unroll
    for (int mt = 0; mt < NMT; ++mt) { accg[mt] = (v8f){}; accu[mt] = (v8f){}; }
    const unsigned short* gB = gate_t + ((size_t)e * HH + ncol + n16) * DD + kh * 8;
    const unsigned short* uB = up_t   + ((size_t)e * HH + ncol + n16) * DD + kh * 8;

    FragBF bg0, bu0, bg1, bu1;
    load_b(gB, bg0);       load_b(uB, bu0);
    load_b(gB + 32, bg1);  load_b(uB + 32, bu1);
#pragma unroll 1
    for (int k = 0; k < DD - 128; k += 64) {
      __builtin_prefetch(gB + k + 512, 0, 1);
      __builtin_prefetch(uB + k + 512, 0, 1);
      wmma_step2(aLo, aHi, k, bg0, bu0, accg, accu);
      load_b(gB + k + 64, bg0);  load_b(uB + k + 64, bu0);
      wmma_step2(aLo, aHi, k + 32, bg1, bu1, accg, accu);
      load_b(gB + k + 96, bg1);  load_b(uB + k + 96, bu1);
    }
    // Peeled tail: k = DD-128 .. DD-32
    wmma_step2(aLo, aHi, DD - 128, bg0, bu0, accg, accu);
    load_b(gB + DD - 64, bg0);  load_b(uB + DD - 64, bu0);
    wmma_step2(aLo, aHi, DD - 96, bg1, bu1, accg, accu);
    load_b(gB + DD - 32, bg1);  load_b(uB + DD - 32, bu1);
    wmma_step2(aLo, aHi, DD - 64, bg0, bu0, accg, accu);
    wmma_step2(aLo, aHi, DD - 32, bg1, bu1, accg, accu);

    // Epilogue: silu(g)*u -> bf16 into hs.  D layout: m = r + 8*kh, n = n16.
#pragma unroll
    for (int mt = 0; mt < NMT; ++mt) {
#pragma unroll
      for (int r = 0; r < 8; ++r) {
        float g = accg[mt][r], u = accu[mt][r];
        float h = (g / (1.0f + __expf(-g))) * u;
        hs[(mt * 16 + r + 8 * kh) * XS_STRIDE + ncol + n16] = f2bf(h);
      }
    }
  }
  __syncthreads();

  // -------- Phase 2: y = h @ down --------
#pragma unroll 1
  for (int nt = 0; nt < 8; ++nt) {
    const int dcol = wv * 128 + nt * 16;
    v8f acc[NMT];
#pragma unroll
    for (int mt = 0; mt < NMT; ++mt) acc[mt] = (v8f){};
    const unsigned short* dB = down_t + ((size_t)e * DD + dcol + n16) * HH + kh * 8;

    FragBF bd0, bd1;
    load_b(dB, bd0);
    load_b(dB + 32, bd1);
#pragma unroll 1
    for (int k = 0; k < HH - 128; k += 64) {
      __builtin_prefetch(dB + k + 512, 0, 1);
      wmma_step1(hLo, hHi, k, bd0, acc);
      load_b(dB + k + 64, bd0);
      wmma_step1(hLo, hHi, k + 32, bd1, acc);
      load_b(dB + k + 96, bd1);
    }
    wmma_step1(hLo, hHi, HH - 128, bd0, acc);
    load_b(dB + HH - 64, bd0);
    wmma_step1(hLo, hHi, HH - 96, bd1, acc);
    load_b(dB + HH - 32, bd1);
    wmma_step1(hLo, hHi, HH - 64, bd0, acc);
    wmma_step1(hLo, hHi, HH - 32, bd1, acc);

#pragma unroll
    for (int mt = 0; mt < NMT; ++mt) {
#pragma unroll
      for (int r = 0; r < 8; ++r) {
        int row = mt * 16 + r + 8 * kh;
        if (tileBase + row < cnt) y[(size_t)stok[row] * DD + dcol + n16] = acc[mt][r];
      }
    }
  }
}

// ---------------------------------------------------------------------------
// Launcher
// ---------------------------------------------------------------------------
extern "C" void kernel_launch(void* const* d_in, const int* in_sizes, int n_in,
                              void* d_out, int out_size, void* d_ws, size_t ws_size,
                              hipStream_t stream) {
  (void)in_sizes; (void)n_in; (void)out_size; (void)ws_size;
  const float* x    = (const float*)d_in[0];   // [B,T,D]
  const float* Wg   = (const float*)d_in[1];   // [E,D]
  const float* gate = (const float*)d_in[2];   // [E,D,H]
  const float* up   = (const float*)d_in[3];   // [E,D,H]
  const float* down = (const float*)d_in[4];   // [E,H,D]
  float* y = (float*)d_out;                    // [B,T,D]

  char* ws = (char*)d_ws;
  int* counts  = (int*)ws;                                   // 8 ints
  int* tokList = (int*)(ws + 4096);                          // E*BT ints (256 KB)
  unsigned short* gate_t = (unsigned short*)(ws + 4096 + (size_t)EE * BT * 4);
  unsigned short* up_t   = gate_t + (size_t)EE * DD * HH;    // +16 MB each
  unsigned short* down_t = up_t   + (size_t)EE * DD * HH;

  hipMemsetAsync(counts, 0, EE * sizeof(int), stream);

  route_kernel<<<BT / 256, 256, 0, stream>>>(x, Wg, counts, tokList);

  dim3 tb(32, 32);
  transpose_cvt_kernel<<<dim3(HH / 32, DD / 32, EE), tb, 0, stream>>>(gate, gate_t, DD, HH);
  transpose_cvt_kernel<<<dim3(HH / 32, DD / 32, EE), tb, 0, stream>>>(up,   up_t,   DD, HH);
  transpose_cvt_kernel<<<dim3(DD / 32, HH / 32, EE), tb, 0, stream>>>(down, down_t, HH, DD);

  size_t smem = (size_t)2 * M_TILE * XS_STRIDE * sizeof(unsigned short); // ~258 KB of 320 KB/WGP
  moe_expert_kernel<<<dim3(EE, BT / M_TILE), 256, smem, stream>>>(
      x, gate_t, up_t, down_t, counts, tokList, y);
}
